// PointNetMSGChosenPReLU_32375463477356
// MI455X (gfx1250) — matrix operand order repaired and checked by reading, not compiled, not run on hardware
//
#include <hip/hip_runtime.h>
#include <hip/hip_bf16.h>

// ---------------------------------------------------------------------------
// PointNet++ MSG set abstraction for MI455X (gfx1250, wave32, WMMA).
// f16 activation streams + f32 WMMA accumulation; BN stats fused into GEMM.
// Wave tile widened along N (NW WMMAs per A-fragment) to cut A-stream HBM
// traffic 2-4x: A is the dominant memory stream in this gather-bound layer.
// ---------------------------------------------------------------------------

typedef __attribute__((ext_vector_type(16))) _Float16 v16h;
typedef __attribute__((ext_vector_type(8)))  float    v8f;
typedef __attribute__((ext_vector_type(8)))  _Float16 h8;

#define BN_EPS 1e-5f

// ---------------- WMMA fragment loads ----------------
// A: 16x32 f16, row-major global. Lane L: m = L%16; half-wave selects K sub-block.
// ISA 16-bit A layout: VGPR0-3 -> K = 8*(L/16)+{0..7}, VGPR4-7 -> K = 16+8*(L/16)+{0..7}.
__device__ __forceinline__ v16h load_a_frag(const _Float16* __restrict__ A, int lda,
                                            int m0, int k0, int lane) {
  const _Float16* row = A + (size_t)(m0 + (lane & 15)) * (size_t)lda;
  const int kh = (lane >> 4) << 3;
  h8 lo = *(const h8*)(row + k0 + kh);        // global_load_b128
  h8 hi = *(const h8*)(row + k0 + 16 + kh);   // global_load_b128
  v16h a;
#pragma unroll
  for (int t = 0; t < 8; ++t) { a[t] = lo[t]; a[8 + t] = hi[t]; }
  return a;
}

// B: 32x16 f16 from LDS tile lB[n][k] (n-major rows of K).
// ISA B layout: elems 0-7 -> K = 8*(L/16)+{0..7}, elems 8-15 -> K = 16+8*(L/16)+{0..7}; n = lane%16.
__device__ __forceinline__ v16h load_b_frag_lds(const _Float16* lB, int K,
                                                int k0, int lane) {
  const _Float16* row = lB + (size_t)(lane & 15) * (size_t)K + k0 + ((lane >> 4) << 3);
  h8 lo = *(const h8*)(row);       // ds_load_b128
  h8 hi = *(const h8*)(row + 16);  // ds_load_b128
  v16h b;
#pragma unroll
  for (int t = 0; t < 8; ++t) { b[t] = lo[t]; b[8 + t] = hi[t]; }
  return b;
}

// ---------------- GEMM: Y[P,N] = X[P,K] * Wt[K,N], + per-channel sum/sum^2 ----
// Block: 128 threads = 4 waves along M. One wave -> one 16 x (16*NW) D tile:
// a single A fragment is reused across NW WMMAs (NW B tiles from LDS).
// P multiple of 64, K multiple of 32, N multiple of 16*NW (guaranteed by caller).
template <int NW>
__global__ void gemm_wmma_stats(const _Float16* __restrict__ A,
                                const _Float16* __restrict__ Wt,
                                _Float16* __restrict__ Y,
                                float* __restrict__ sums,
                                float* __restrict__ sumsq,
                                int P, int K, int N) {
  extern __shared__ _Float16 lB[];   // [16*NW][K], n-major
  const int lane = threadIdx.x & 31;
  const int wave = threadIdx.x >> 5;
  const int n0 = blockIdx.y * (16 * NW);

  // Stage W^T column tile into LDS: lB[n*K + k] = Wt[k*N + n0+n]
  for (int i = threadIdx.x; i < K * 16 * NW; i += blockDim.x) {
    const int n = i / K;
    const int k = i - n * K;
    lB[i] = Wt[(size_t)k * (size_t)N + (size_t)(n0 + n)];
  }
  __syncthreads();

  const int m0 = ((blockIdx.x << 2) + wave) << 4;
  if (m0 >= P) return;               // wave-uniform: EXEC stays all-ones for WMMA

  v8f c[NW];
#pragma unroll
  for (int w = 0; w < NW; ++w) c[w] = (v8f){};

  for (int k0 = 0; k0 < K; k0 += 32) {
    v16h a = load_a_frag(A, K, m0, k0, lane);
#pragma unroll
    for (int w = 0; w < NW; ++w) {
      v16h b = load_b_frag_lds(lB + (size_t)(w * 16) * (size_t)K, K, k0, lane);
      c[w] = __builtin_amdgcn_wmma_f32_16x16x32_f16(false, a, false, b, (short)0,
                                                    c[w], false, false);
    }
  }

  // D layout: element v -> m = m0 + 8*(lane/16) + v, n = ntile + lane%16
  const int mb = m0 + ((lane >> 4) << 3);
#pragma unroll
  for (int w = 0; w < NW; ++w) {
    const int n = n0 + w * 16 + (lane & 15);
    float s = 0.f, s2 = 0.f;
#pragma unroll
    for (int v = 0; v < 8; ++v) {
      const float y = c[w][v];
      Y[(size_t)(mb + v) * (size_t)N + (size_t)n] = (_Float16)y;
      s += y; s2 += y * y;
    }
    // lanes L and L+16 share the same column n -> pairwise reduce, then atomics
    s  += __shfl_down(s, 16, 32);
    s2 += __shfl_down(s2, 16, 32);
    if (lane < 16) {
      atomicAdd(&sums[n], s);
      atomicAdd(&sumsq[n], s2);
    }
  }
}

// ---------------- Ball query: one wave per query point -----------------------
// First-K-in-ascending-index semantics via ballot-ordered compaction.
__global__ void ball_query(const float* __restrict__ xyz,
                           const float* __restrict__ nxyz,
                           int* __restrict__ idx0, int* __restrict__ idx1,
                           int* __restrict__ idx2, int B, int N, int S) {
  const int q    = (blockIdx.x * blockDim.x + threadIdx.x) >> 5;
  const int lane = threadIdx.x & 31;
  if (q >= B * S) return;
  const int b = q / S;
  const float qx = nxyz[(size_t)q * 3 + 0];
  const float qy = nxyz[(size_t)q * 3 + 1];
  const float qz = nxyz[(size_t)q * 3 + 2];
  const float* px = xyz + (size_t)b * N * 3;

  const float r2[3] = {0.01f, 0.04f, 0.16f};
  const int   Ks[3] = {16, 32, 64};
  int* outp[3] = {idx0 + (size_t)q * 16, idx1 + (size_t)q * 32, idx2 + (size_t)q * 64};
  int cnt[3]   = {0, 0, 0};
  int first[3] = {N - 1, N - 1, N - 1};

  for (int nb = 0; nb < N; nb += 32) {
    const int n = nb + lane;
    const float dx = px[(size_t)n * 3 + 0] - qx;
    const float dy = px[(size_t)n * 3 + 1] - qy;
    const float dz = px[(size_t)n * 3 + 2] - qz;
    const float d2 = dx * dx + dy * dy + dz * dz;
#pragma unroll
    for (int sc = 0; sc < 3; ++sc) {
      if (cnt[sc] >= Ks[sc]) continue;                       // wave-uniform
      const unsigned mask = (unsigned)__ballot(d2 <= r2[sc]);
      if (mask == 0u) continue;
      if (cnt[sc] == 0) first[sc] = nb + (__ffs(mask) - 1);
      if (d2 <= r2[sc]) {
        const int rank = __popc(mask & ((1u << lane) - 1u));
        const int slot = cnt[sc] + rank;
        if (slot < Ks[sc]) outp[sc][slot] = n;
      }
      cnt[sc] += __popc(mask);
    }
    if (cnt[0] >= 16 && cnt[1] >= 32 && cnt[2] >= 64) break;
  }
#pragma unroll
  for (int sc = 0; sc < 3; ++sc) {
    const int c = cnt[sc] > Ks[sc] ? Ks[sc] : cnt[sc];
    for (int j = c + lane; j < Ks[sc]; j += 32) outp[sc][j] = first[sc];
  }
}

// ---------------- Gather grouped features + recentred xyz -> f16 [P, 96] -----
__global__ void gather_group(const float* __restrict__ feats,
                             const float* __restrict__ xyz,
                             const float* __restrict__ nxyz,
                             const int* __restrict__ idx,
                             _Float16* __restrict__ X,
                             int N, int S, int K, int Cpad, int total) {
  const int t = blockIdx.x * blockDim.x + threadIdx.x;
  if (t >= total) return;
  const int p = t / Cpad, c = t - p * Cpad;
  const int q = p / K;                 // = b*S + s
  const int b = q / S;
  const int id = idx[p];
  float v;
  if (c < 64) {
    v = feats[((size_t)b * N + id) * 64 + c];
  } else if (c < 67) {
    const int d = c - 64;
    v = xyz[((size_t)b * N + id) * 3 + d] - nxyz[(size_t)q * 3 + d];
  } else {
    v = 0.f;                           // zero padding (dead under W padding too)
  }
  X[(size_t)p * Cpad + c] = (_Float16)v;
}

// ---------------- Weight convert: W[cout,cin] f32 -> Wt[cinpad,cout] f16 -----
__global__ void cvt_weight(const float* __restrict__ W, _Float16* __restrict__ Wt,
                           int cout, int cin, int cinpad) {
  const int t = blockIdx.x * blockDim.x + threadIdx.x;
  if (t >= cinpad * cout) return;
  const int k = t / cout, n = t - k * cout;
  const float v = (k < cin) ? W[(size_t)n * cin + k] : 0.f;
  Wt[(size_t)k * cout + n] = (_Float16)v;
}

// ---------------- BN helpers -------------------------------------------------
__global__ void zero_stats(float* a, float* b, int n) {
  const int t = blockIdx.x * blockDim.x + threadIdx.x;
  if (t < n) { a[t] = 0.f; b[t] = 0.f; }
}

__global__ void bn_finalize(const float* __restrict__ sums,
                            const float* __restrict__ sumsq,
                            const float* __restrict__ gamma,
                            const float* __restrict__ beta,
                            float* __restrict__ bnscale,
                            float* __restrict__ bnshift, int N, float invP) {
  const int c = blockIdx.x * blockDim.x + threadIdx.x;
  if (c >= N) return;
  const float mean = sums[c] * invP;
  const float var  = sumsq[c] * invP - mean * mean;
  const float sc   = gamma[c] * rsqrtf(var + BN_EPS);
  bnscale[c] = sc;
  bnshift[c] = beta[c] - mean * sc;
}

// BN + PReLU elementwise: f16 in -> f16 out (next layer input)
__global__ void bn_act(const _Float16* __restrict__ Y,
                       const float* __restrict__ bnscale,
                       const float* __restrict__ bnshift,
                       const float* __restrict__ alpha,
                       _Float16* __restrict__ Xn, int total, int N) {
  const int t = blockIdx.x * blockDim.x + threadIdx.x;
  if (t >= total) return;
  const int c = t % N;
  float y = (float)Y[t] * bnscale[c] + bnshift[c];
  y = y > 0.f ? y : alpha[c] * y;
  Xn[t] = (_Float16)y;
}

// Fused BN + PReLU + max over K neighbors -> pooled[q, coff + c]
__global__ void maxpool_bn_act(const _Float16* __restrict__ Y,
                               const float* __restrict__ bnscale,
                               const float* __restrict__ bnshift,
                               const float* __restrict__ alpha,
                               _Float16* __restrict__ pooled,
                               int Q, int K, int N, int coff, int pstride) {
  const int t = blockIdx.x * blockDim.x + threadIdx.x;
  if (t >= Q * N) return;
  const int q = t / N, c = t - q * N;
  const float sc = bnscale[c], sh = bnshift[c], al = alpha[c];
  float m = -3.4e38f;
  for (int j = 0; j < K; ++j) {
    float y = (float)Y[((size_t)q * K + j) * (size_t)N + c] * sc + sh;
    y = y > 0.f ? y : al * y;
    m = fmaxf(m, y);
  }
  pooled[(size_t)q * pstride + coff + c] = (_Float16)m;
}

// Final BN + PReLU -> f32 output
__global__ void final_out(const _Float16* __restrict__ Y,
                          const float* __restrict__ bnscale,
                          const float* __restrict__ bnshift,
                          const float* __restrict__ alpha,
                          float* __restrict__ out, int total, int N) {
  const int t = blockIdx.x * blockDim.x + threadIdx.x;
  if (t >= total) return;
  const int c = t % N;
  float y = (float)Y[t] * bnscale[c] + bnshift[c];
  out[t] = y > 0.f ? y : alpha[c] * y;
}

__global__ void copy_f32(const float* __restrict__ src, float* __restrict__ dst, int n) {
  const int t = blockIdx.x * blockDim.x + threadIdx.x;
  if (t < n) dst[t] = src[t];
}

// ---------------------------------------------------------------------------
// Host orchestration
// ---------------------------------------------------------------------------
struct LayerDesc { int cin, cinpad, cout; };

static inline void launch_gemm(const _Float16* A, const _Float16* Wt, _Float16* Y,
                               float* ssum, float* ssq, int P, int K, int N,
                               hipStream_t stream) {
  // Prefer 16x64 wave tiles (4x A reuse) when N allows; all couts are mult of 32.
  if (N % 64 == 0) {
    dim3 grid(P / 64, N / 64);
    gemm_wmma_stats<4><<<grid, 128, (size_t)K * 64 * sizeof(_Float16), stream>>>(
        A, Wt, Y, ssum, ssq, P, K, N);
  } else {
    dim3 grid(P / 64, N / 32);
    gemm_wmma_stats<2><<<grid, 128, (size_t)K * 32 * sizeof(_Float16), stream>>>(
        A, Wt, Y, ssum, ssq, P, K, N);
  }
}

extern "C" void kernel_launch(void* const* d_in, const int* in_sizes, int n_in,
                              void* d_out, int out_size, void* d_ws, size_t ws_size,
                              hipStream_t stream) {
  (void)in_sizes; (void)n_in; (void)out_size; (void)ws_size;

  const int B = 4, N = 8192, S = 2048;
  const int Q = B * S;                       // 8192 query points
  const int Ks[3]   = {16, 32, 64};
  const int coff[3] = {0, 64, 192};          // concat offsets (64+128+128=320)

  static const LayerDesc layers[10] = {
    {67, 96, 32}, {32, 32, 32}, {32, 32, 64},     // block 0
    {67, 96, 64}, {64, 64, 64}, {64, 64, 128},    // block 1
    {67, 96, 64}, {64, 64, 96}, {96, 96, 128},    // block 2
    {320, 320, 256}                               // final
  };

  const float* xyz   = (const float*)d_in[0];
  const float* nxyz  = (const float*)d_in[1];
  const float* feats = (const float*)d_in[2];
  // Param leaves per layer (JAX dict key order: alpha, b, beta, gamma, w):
  // d_in[3 + l*5 + {0:alpha, 1:b(dead, BN cancels it), 2:beta, 3:gamma, 4:w}]

  // ---- workspace carve-up ----
  char* ws = (char*)d_ws;
  size_t off = 0;
  auto take = [&](size_t bytes) -> void* {
    void* p = ws + off;
    off = (off + bytes + 255) & ~(size_t)255;
    return p;
  };
  int* idxs[3];
  for (int s = 0; s < 3; ++s) idxs[s] = (int*)take((size_t)Q * Ks[s] * sizeof(int));
  const size_t Pmax = (size_t)Q * 64;                       // 524288
  _Float16* Xbuf   = (_Float16*)take(Pmax * 96  * sizeof(_Float16));
  _Float16* Ybuf   = (_Float16*)take(Pmax * 128 * sizeof(_Float16));
  _Float16* pooled = (_Float16*)take((size_t)Q * 320 * sizeof(_Float16));
  float* ssum    = (float*)take(256 * sizeof(float));
  float* ssq     = (float*)take(256 * sizeof(float));
  float* bnscale = (float*)take(256 * sizeof(float));
  float* bnshift = (float*)take(256 * sizeof(float));
  _Float16* Wt[10];
  for (int l = 0; l < 10; ++l)
    Wt[l] = (_Float16*)take((size_t)layers[l].cinpad * layers[l].cout * sizeof(_Float16));

  // ---- 1. convert / pad / transpose weights to f16 ----
  for (int l = 0; l < 10; ++l) {
    const float* w = (const float*)d_in[3 + l * 5 + 4];
    const int tot = layers[l].cinpad * layers[l].cout;
    cvt_weight<<<(tot + 255) / 256, 256, 0, stream>>>(w, Wt[l], layers[l].cout,
                                                      layers[l].cin, layers[l].cinpad);
  }

  // ---- 2. ball query (one wave per query) + output new_xyz passthrough ----
  ball_query<<<(Q * 32 + 255) / 256, 256, 0, stream>>>(xyz, nxyz, idxs[0], idxs[1],
                                                       idxs[2], B, N, S);
  copy_f32<<<(Q * 3 + 255) / 256, 256, 0, stream>>>(nxyz, (float*)d_out, Q * 3);

  // ---- 3. per-scale grouped MLP ----
  for (int s = 0; s < 3; ++s) {
    const int K = Ks[s];
    const int P = Q * K;                      // rows; multiple of 64
    const int gtot = P * 96;
    gather_group<<<(gtot + 255) / 256, 256, 0, stream>>>(feats, xyz, nxyz, idxs[s],
                                                         Xbuf, N, S, K, 96, gtot);
    for (int li = 0; li < 3; ++li) {
      const int l = 3 * s + li;
      const LayerDesc& L = layers[l];
      const float* alpha = (const float*)d_in[3 + l * 5 + 0];
      const float* beta  = (const float*)d_in[3 + l * 5 + 2];
      const float* gamma = (const float*)d_in[3 + l * 5 + 3];
      zero_stats<<<1, 256, 0, stream>>>(ssum, ssq, 256);
      launch_gemm(Xbuf, Wt[l], Ybuf, ssum, ssq, P, L.cinpad, L.cout, stream);
      bn_finalize<<<1, 256, 0, stream>>>(ssum, ssq, gamma, beta, bnscale, bnshift,
                                         L.cout, 1.0f / (float)P);
      if (li < 2) {
        const int t2 = P * L.cout;
        bn_act<<<(t2 + 255) / 256, 256, 0, stream>>>(Ybuf, bnscale, bnshift, alpha,
                                                     Xbuf, t2, L.cout);
      } else {
        const int t3 = Q * L.cout;
        maxpool_bn_act<<<(t3 + 255) / 256, 256, 0, stream>>>(
            Ybuf, bnscale, bnshift, alpha, pooled, Q, K, L.cout, coff[s], 320);
      }
    }
  }

  // ---- 4. final 320 -> 256 layer over Q points ----
  {
    const LayerDesc& L = layers[9];
    const float* alpha = (const float*)d_in[3 + 9 * 5 + 0];
    const float* beta  = (const float*)d_in[3 + 9 * 5 + 2];
    const float* gamma = (const float*)d_in[3 + 9 * 5 + 3];
    zero_stats<<<1, 256, 0, stream>>>(ssum, ssq, 256);
    launch_gemm(pooled, Wt[9], Ybuf, ssum, ssq, Q, L.cinpad, L.cout, stream);
    bn_finalize<<<1, 256, 0, stream>>>(ssum, ssq, gamma, beta, bnscale, bnshift,
                                       L.cout, 1.0f / (float)Q);
    const int t4 = Q * L.cout;
    final_out<<<(t4 + 255) / 256, 256, 0, stream>>>(Ybuf, bnscale, bnshift, alpha,
                                                    (float*)d_out + Q * 3, t4, L.cout);
  }
}
